// TransformerBlock_18811956756975
// MI455X (gfx1250) — matrix operand order repaired and checked by reading, not compiled, not run on hardware
//
#include <hip/hip_runtime.h>
#include <hip/hip_bf16.h>

typedef __attribute__((ext_vector_type(16))) __bf16 v16bf;
typedef __attribute__((ext_vector_type(8)))  float  v8f;

// ---------------- bf16 helpers (bit-level, no __bf16 arithmetic) ----------------
__device__ __forceinline__ float bf2f(__bf16 x) {
    unsigned short u = __builtin_bit_cast(unsigned short, x);
    unsigned v = (unsigned)u << 16;
    return __builtin_bit_cast(float, v);
}
__device__ __forceinline__ __bf16 f2bf(float f) {
    unsigned u = __builtin_bit_cast(unsigned, f);
    u += 0x7fffu + ((u >> 16) & 1u);                 // round-to-nearest-even
    unsigned short s = (unsigned short)(u >> 16);
    return __builtin_bit_cast(__bf16, s);
}
// two 16-byte vector loads -> one v16bf WMMA operand
__device__ __forceinline__ v16bf load2x16(const __bf16* p0, const __bf16* p1) {
    union U { uint4 q[2]; v16bf v; } u;
    u.q[0] = *(const uint4*)p0;
    u.q[1] = *(const uint4*)p1;
    return u.v;
}

// ---------------- f32 -> bf16 convert (weights) ----------------
__global__ void cvt_f32_bf16_kernel(const float* __restrict__ in, __bf16* __restrict__ out, int n) {
    int i = blockIdx.x * blockDim.x + threadIdx.x;
    if (i < n) out[i] = f2bf(in[i]);
}

// ---------------- permute k_w [co][ci][3][3] -> bf16 [co][tap][ci] ----------------
__global__ void permute_kw_kernel(const float* __restrict__ in, __bf16* __restrict__ out, int C) {
    int i = blockIdx.x * blockDim.x + threadIdx.x;      // over C*C*9
    if (i >= C * C * 9) return;
    int co = i / (C * 9);
    int rem = i - co * C * 9;
    int t = rem / C;
    int ci = rem - t * C;
    out[i] = f2bf(in[(size_t)co * C * 9 + ci * 9 + t]);
}

// ---------------- LN1: read x NCHW, write NHWC bf16 ----------------
__global__ void ln1_kernel(const float* __restrict__ X, const float* __restrict__ g,
                           const float* __restrict__ bta, __bf16* __restrict__ outT,
                           int C, int HW) {
    int b = blockIdx.z;
    int p = blockIdx.x * blockDim.x + threadIdx.x;
    if (p >= HW) return;
    const float* xp = X + (size_t)b * C * HW + p;
    float s = 0.f, ss = 0.f;
    for (int c = 0; c < C; ++c) { float v = xp[(size_t)c * HW]; s += v; ss += v * v; }
    float mean = s / (float)C;
    float var  = ss / (float)C - mean * mean;
    float rstd = rsqrtf(var + 1e-5f);
    __bf16* op = outT + ((size_t)b * HW + p) * C;
    for (int c = 0; c < C; ++c)
        op[c] = f2bf((xp[(size_t)c * HW] - mean) * rstd * g[c] + bta[c]);
}

// ---------------- LN2: read NHWC f32, write NHWC bf16 ----------------
__global__ void ln2_kernel(const float* __restrict__ XT, const float* __restrict__ g,
                           const float* __restrict__ bta, __bf16* __restrict__ outT,
                           int C, int HW) {
    int b = blockIdx.z;
    int p = blockIdx.x * blockDim.x + threadIdx.x;
    if (p >= HW) return;
    const float* xp = XT + ((size_t)b * HW + p) * C;
    float s = 0.f, ss = 0.f;
    for (int c = 0; c < C; ++c) { float v = xp[c]; s += v; ss += v * v; }
    float mean = s / (float)C;
    float var  = ss / (float)C - mean * mean;
    float rstd = rsqrtf(var + 1e-5f);
    __bf16* op = outT + ((size_t)b * HW + p) * C;
    for (int c = 0; c < C; ++c)
        op[c] = f2bf((xp[c] - mean) * rstd * g[c] + bta[c]);
}

// NT-wide WMMA row: compile-time unrolled so reuse_a is a constant (ICE required).
template <int NT>
__device__ __forceinline__ void wmma_row(const v16bf& at, const v16bf (&bt)[NT], v8f (&acc)[NT]) {
    static_assert(NT >= 1 && NT <= 2, "NT must be 1 or 2");
    acc[0] = __builtin_amdgcn_wmma_f32_16x16x32_bf16(
        false, at, false, bt[0], (short)0, acc[0], /*reuse_a=*/false, /*reuse_b=*/false);
    if constexpr (NT == 2) {
        acc[1] = __builtin_amdgcn_wmma_f32_16x16x32_bf16(
            false, at, false, bt[1], (short)0, acc[1], /*reuse_a=*/true, /*reuse_b=*/false);
    }
}

// ---------------- generic bf16 WMMA GEMM ----------------
// A: [M][K] row-major bf16.  B: [b][N][K] pixel-major bf16 (B[k][n] = Brow(n)[k]).
// Each wave: MT cout-tiles x NT pixel-tiles; A tile regs reused across NT (reuse_a hint).
// EPI: 0 = f32 NCHW out                       (a_pw)
//      1 = f32 NHWC out, * aux(NCHW gate)     (v, illu gate)
//      2 = f32 NHWC out, + aux(NCHW residual) (po, +x)
//      3 = bf16 NHWC out                      (pw1)
//      4 = f32 NCHW out, + aux(NHWC residual) (pw2 final, +x2)
template <int MT, int NT, int EPI>
__global__ void gemm_bf16_kernel(const __bf16* __restrict__ A,
                                 const __bf16* __restrict__ Bm,
                                 const float* __restrict__ bias,
                                 float* __restrict__ DoutF,
                                 __bf16* __restrict__ DoutB,
                                 const float* __restrict__ aux,
                                 int M, int N, int K) {
    int lane = threadIdx.x & 31;
    int wave = threadIdx.x >> 5;
    int b = blockIdx.z;
    int ntile = (blockIdx.x * (blockDim.x >> 5) + wave) * (16 * NT);
    int m0 = blockIdx.y * (16 * MT);
    if (ntile >= N) return;                       // wave-uniform
    int nl = lane & 15;
    int kh = lane >> 4;
    const __bf16* Brow[NT];
    #pragma unroll
    for (int nt = 0; nt < NT; ++nt)
        Brow[nt] = Bm + ((size_t)b * N + ntile + nt * 16 + nl) * K;

    v8f zero = {0.f,0.f,0.f,0.f,0.f,0.f,0.f,0.f};
    v8f acc[MT][NT];
    #pragma unroll
    for (int i = 0; i < MT; ++i)
        #pragma unroll
        for (int j = 0; j < NT; ++j) acc[i][j] = zero;

    for (int kb = 0; kb < K; kb += 32) {
        v16bf bt[NT];
        #pragma unroll
        for (int nt = 0; nt < NT; ++nt) {
            const __bf16* bp = Brow[nt] + kb + kh * 16;
            bt[nt] = load2x16(bp, bp + 8);
            if (kb + 32 < K) __builtin_prefetch(bp + 32, 0, 1);   // global_prefetch_b8
        }
        #pragma unroll
        for (int mt = 0; mt < MT; ++mt) {
            int m = m0 + mt * 16 + nl;
            const __bf16* ap = A + (size_t)m * K + kb + kh * 16;
            v16bf at = load2x16(ap, ap + 8);
            wmma_row<NT>(at, bt, acc[mt]);
        }
    }
    // D layout: VGPR r -> m = mbase + r (mbase = m0 + mt*16 + 8*kh), fixed n per lane
    #pragma unroll
    for (int mt = 0; mt < MT; ++mt) {
        int mbase = m0 + mt * 16 + 8 * kh;
        #pragma unroll
        for (int nt = 0; nt < NT; ++nt) {
            int n = ntile + nt * 16 + nl;
            if (EPI == 0) {
                #pragma unroll
                for (int r = 0; r < 8; ++r)
                    DoutF[((size_t)b * M + mbase + r) * N + n] = acc[mt][nt][r] + bias[mbase + r];
            } else if (EPI == 1 || EPI == 2) {
                float tmp[8];
                #pragma unroll
                for (int r = 0; r < 8; ++r) {
                    float val = acc[mt][nt][r] + bias[mbase + r];
                    float ax = aux[((size_t)b * M + mbase + r) * N + n];
                    tmp[r] = (EPI == 1) ? val * ax : val + ax;
                }
                float* dp = DoutF + ((size_t)b * N + n) * M + mbase;
                *(float4*)(dp)     = make_float4(tmp[0], tmp[1], tmp[2], tmp[3]);
                *(float4*)(dp + 4) = make_float4(tmp[4], tmp[5], tmp[6], tmp[7]);
            } else if (EPI == 3) {
                __bf16 t8[8];
                #pragma unroll
                for (int r = 0; r < 8; ++r) t8[r] = f2bf(acc[mt][nt][r] + bias[mbase + r]);
                *(uint4*)(DoutB + ((size_t)b * N + n) * M + mbase) = *(uint4*)t8;
            } else { // EPI == 4
                const float* ar = aux + ((size_t)b * N + n) * M + mbase;
                #pragma unroll
                for (int r = 0; r < 8; ++r)
                    DoutF[((size_t)b * M + mbase + r) * N + n] =
                        acc[mt][nt][r] + bias[mbase + r] + ar[r];
            }
        }
    }
}

// ---------------- implicit-GEMM 3x3 s2 pad1 dense conv via WMMA ----------------
// A: [M][K=9*C] tap-major weights.  Xn: [b][Hin*Win][C] NHWC bf16.  Dout: [b][M][N] NCHW f32.
template <int MT, int NT>
__global__ void conv3x3s2_wmma_kernel(const __bf16* __restrict__ A,
                                      const __bf16* __restrict__ XnT,
                                      const float* __restrict__ bias,
                                      float* __restrict__ Dout,
                                      int M, int C, int Hin, int Win, int Hout, int Wout) {
    int lane = threadIdx.x & 31;
    int wave = threadIdx.x >> 5;
    int b = blockIdx.z;
    int N = Hout * Wout;
    int K = C * 9;
    int ntile = (blockIdx.x * (blockDim.x >> 5) + wave) * (16 * NT);
    int m0 = blockIdx.y * (16 * MT);
    if (ntile >= N) return;
    int nl = lane & 15;
    int kh = lane >> 4;
    int oh[NT], ow[NT];
    #pragma unroll
    for (int nt = 0; nt < NT; ++nt) {
        int n = ntile + nt * 16 + nl;
        oh[nt] = n / Wout; ow[nt] = n % Wout;
    }
    const __bf16* Xb = XnT + (size_t)b * Hin * Win * C;
    v8f zero = {0.f,0.f,0.f,0.f,0.f,0.f,0.f,0.f};
    v8f acc[MT][NT];
    #pragma unroll
    for (int i = 0; i < MT; ++i)
        #pragma unroll
        for (int j = 0; j < NT; ++j) acc[i][j] = zero;

    for (int kb = 0; kb < K; kb += 32) {
        int kbase = kb + kh * 16;
        int t  = kbase / 128;           // C == 128: tap index (16-runs never cross taps)
        int ci = kbase - t * 128;
        int dh = t / 3 - 1, dw = t % 3 - 1;
        v16bf bt[NT];
        #pragma unroll
        for (int nt = 0; nt < NT; ++nt) {
            int ih = oh[nt] * 2 + dh;
            int iw = ow[nt] * 2 + dw;
            if (ih >= 0 && ih < Hin && iw >= 0 && iw < Win) {
                const __bf16* bp = Xb + ((size_t)ih * Win + iw) * C + ci;
                bt[nt] = load2x16(bp, bp + 8);
            } else {
                #pragma unroll
                for (int e = 0; e < 16; ++e)
                    bt[nt][e] = __builtin_bit_cast(__bf16, (unsigned short)0);
            }
        }
        #pragma unroll
        for (int mt = 0; mt < MT; ++mt) {
            int m = m0 + mt * 16 + nl;
            const __bf16* ap = A + (size_t)m * K + kb + kh * 16;
            v16bf at = load2x16(ap, ap + 8);
            wmma_row<NT>(at, bt, acc[mt]);
        }
    }
    #pragma unroll
    for (int mt = 0; mt < MT; ++mt) {
        int mbase = m0 + mt * 16 + 8 * kh;
        #pragma unroll
        for (int nt = 0; nt < NT; ++nt) {
            int n = ntile + nt * 16 + nl;
            #pragma unroll
            for (int r = 0; r < 8; ++r)
                Dout[((size_t)b * M + mbase + r) * N + n] = acc[mt][nt][r] + bias[mbase + r];
        }
    }
}

// ---------------- depthwise 3x3 s2 pad1, NHWC bf16 in; out channel-major f32 or NHWC bf16 ----
__global__ void dwconv3x3s2_kernel(const __bf16* __restrict__ XT, const float* __restrict__ W9,
                                   const float* __restrict__ bias,
                                   float* __restrict__ outCM, __bf16* __restrict__ outPM,
                                   int C, int Hin, int Win, int Hout, int Wout) {
    int b = blockIdx.z;
    int p = blockIdx.x;                 // output pixel
    int c = threadIdx.x;                // channel
    int oh = p / Wout, ow = p % Wout;
    const __bf16* xb = XT + (size_t)b * Hin * Win * C;
    float acc = bias[c];
    #pragma unroll
    for (int r = 0; r < 3; ++r) {
        int ih = oh * 2 - 1 + r;
        if (ih < 0 || ih >= Hin) continue;
        #pragma unroll
        for (int s = 0; s < 3; ++s) {
            int iw = ow * 2 - 1 + s;
            if (iw < 0 || iw >= Win) continue;
            acc += W9[c * 9 + r * 3 + s] * bf2f(xb[((size_t)ih * Win + iw) * C + c]);
        }
    }
    if (outCM) outCM[((size_t)b * C + c) * (Hout * Wout) + p] = acc;
    else       outPM[((size_t)b * (Hout * Wout) + p) * C + c] = f2bf(acc);
}

// ---------------- depthwise 3x3 s1 pad1 + SiLU, NHWC bf16 in/out ----------------
__global__ void dwconv3x3_silu_kernel(const __bf16* __restrict__ XT, const float* __restrict__ W9,
                                      const float* __restrict__ bias, __bf16* __restrict__ outT,
                                      int C, int H, int Wd) {
    int b = blockIdx.z;
    int p = blockIdx.x;
    int c = threadIdx.x;
    int oh = p / Wd, ow = p % Wd;
    const __bf16* xb = XT + (size_t)b * H * Wd * C;
    float acc = bias[c];
    #pragma unroll
    for (int r = 0; r < 3; ++r) {
        int ih = oh - 1 + r;
        if (ih < 0 || ih >= H) continue;
        #pragma unroll
        for (int s = 0; s < 3; ++s) {
            int iw = ow - 1 + s;
            if (iw < 0 || iw >= Wd) continue;
            acc += W9[c * 9 + r * 3 + s] * bf2f(xb[((size_t)ih * Wd + iw) * C + c]);
        }
    }
    float sil = acc / (1.f + __expf(-acc));
    outT[((size_t)b * H * Wd + p) * C + c] = f2bf(sil);
}

// ---------------- per-row inverse L2 norm (channel-major rows) ----------------
__global__ void rownorm_kernel(const float* __restrict__ X, float* __restrict__ inv, int n) {
    __shared__ float red[256];
    int row = blockIdx.x;
    const float* xp = X + (size_t)row * n;
    float ss = 0.f;
    for (int i = threadIdx.x; i < n; i += blockDim.x) { float v = xp[i]; ss += v * v; }
    red[threadIdx.x] = ss; __syncthreads();
    for (int st = 128; st > 0; st >>= 1) {
        if ((int)threadIdx.x < st) red[threadIdx.x] += red[threadIdx.x + st];
        __syncthreads();
    }
    if (threadIdx.x == 0) inv[row] = 1.f / fmaxf(sqrtf(red[0]), 1e-12f);
}

// ---------------- Gram + softmax + fused M = attn_a @ attn_k per (b,h) ----------------
__global__ void attn_gram_kernel(const float* __restrict__ q, const float* __restrict__ k,
                                 const float* __restrict__ a,
                                 const float* __restrict__ invq, const float* __restrict__ invk,
                                 const float* __restrict__ inva,
                                 const float* __restrict__ tempA, const float* __restrict__ tempV,
                                 float* __restrict__ Mout, int n) {
    int b = blockIdx.z, h = blockIdx.y;
    __shared__ float red[256];
    __shared__ float Gqa[16][8];
    __shared__ float Gak[8][16];
    __shared__ float Aa[16][8];
    __shared__ float Ak[8][16];
    const float* qb = q + ((size_t)b * 128 + h * 16) * n;
    const float* kb = k + ((size_t)b * 128 + h * 16) * n;
    const float* ab = a + ((size_t)b * 64  + h * 8 ) * n;
    for (int pair = 0; pair < 256; ++pair) {
        const float *r0, *r1;
        if (pair < 128) { int c = pair >> 3, d = pair & 7; r0 = qb + (size_t)c * n; r1 = ab + (size_t)d * n; }
        else { int t = pair - 128; int d = t >> 4, c = t & 15; r0 = ab + (size_t)d * n; r1 = kb + (size_t)c * n; }
        float s = 0.f;
        for (int i = threadIdx.x; i < n; i += 256) s += r0[i] * r1[i];
        red[threadIdx.x] = s; __syncthreads();
        for (int st = 128; st > 0; st >>= 1) {
            if ((int)threadIdx.x < st) red[threadIdx.x] += red[threadIdx.x + st];
            __syncthreads();
        }
        if (threadIdx.x == 0) {
            if (pair < 128) Gqa[pair >> 3][pair & 7] = red[0];
            else { int t = pair - 128; Gak[t >> 4][t & 15] = red[0]; }
        }
        __syncthreads();
    }
    float ta = tempA[h], tv = tempV[h];
    if (threadIdx.x < 16) {
        int c = threadIdx.x;
        float iq = invq[b * 128 + h * 16 + c];
        float mx = -1e30f, tmp[8];
        for (int d = 0; d < 8; ++d) {
            float v = Gqa[c][d] * iq * inva[b * 64 + h * 8 + d] * ta;
            tmp[d] = v; mx = fmaxf(mx, v);
        }
        float se = 0.f;
        for (int d = 0; d < 8; ++d) { tmp[d] = __expf(tmp[d] - mx); se += tmp[d]; }
        for (int d = 0; d < 8; ++d) Aa[c][d] = tmp[d] / se;
    } else if (threadIdx.x >= 32 && threadIdx.x < 40) {
        int d = threadIdx.x - 32;
        float ia = inva[b * 64 + h * 8 + d];
        float mx = -1e30f, tmp[16];
        for (int c = 0; c < 16; ++c) {
            float v = Gak[d][c] * ia * invk[b * 128 + h * 16 + c] * tv;
            tmp[c] = v; mx = fmaxf(mx, v);
        }
        float se = 0.f;
        for (int c = 0; c < 16; ++c) { tmp[c] = __expf(tmp[c] - mx); se += tmp[c]; }
        for (int c = 0; c < 16; ++c) Ak[d][c] = tmp[c] / se;
    }
    __syncthreads();
    {   // M[c][cc] = sum_d Aa[c][d] * Ak[d][cc]
        int c = threadIdx.x >> 4, cc = threadIdx.x & 15;
        float s = 0.f;
        for (int d = 0; d < 8; ++d) s += Aa[c][d] * Ak[d][cc];
        Mout[(size_t)(b * 8 + h) * 256 + c * 16 + cc] = s;
    }
}

// ---------------- out = M(16x16) @ v per (b,h); v,out pixel-major ----------------
__global__ void attn_apply_kernel(const float* __restrict__ Mmat, const float* __restrict__ vT,
                                  __bf16* __restrict__ outT, int HW) {
    int b = blockIdx.z, h = blockIdx.y;
    __shared__ float Ms[256];
    Ms[threadIdx.x] = Mmat[(size_t)(b * 8 + h) * 256 + threadIdx.x];
    __syncthreads();
    int p = blockIdx.x * blockDim.x + threadIdx.x;
    if (p >= HW) return;
    const float* vp = vT + ((size_t)b * HW + p) * 128 + h * 16;
    float vin[16];
    #pragma unroll
    for (int d = 0; d < 16; ++d) vin[d] = vp[d];
    __bf16 t16[16];
    #pragma unroll
    for (int c = 0; c < 16; ++c) {
        float s = 0.f;
        #pragma unroll
        for (int d = 0; d < 16; ++d) s += Ms[c * 16 + d] * vin[d];
        t16[c] = f2bf(s);
    }
    __bf16* op = outT + ((size_t)b * HW + p) * 128 + h * 16;
    *(uint4*)(op)     = *(uint4*)(t16);
    *(uint4*)(op + 8) = *(uint4*)(t16 + 8);
}

// ===================================================================================
extern "C" void kernel_launch(void* const* d_in, const int* in_sizes, int n_in,
                              void* d_out, int out_size, void* d_ws, size_t ws_size,
                              hipStream_t stream) {
    (void)in_sizes; (void)n_in; (void)out_size; (void)ws_size;

    const int B = 2, C = 128, H = 256, W = 256, MID = 256;
    const int HW = H * W;                         // 65536
    const int H2 = 128, W2 = 128, P2 = H2 * W2;   // 16384

    const float* x      = (const float*)d_in[0];
    const float* illu   = (const float*)d_in[1];
    const float* ln1_w  = (const float*)d_in[2];
    const float* ln1_b  = (const float*)d_in[3];
    const float* q_w    = (const float*)d_in[4];
    const float* q_b    = (const float*)d_in[5];
    const float* k_w    = (const float*)d_in[6];
    const float* k_b    = (const float*)d_in[7];
    const float* adw_w  = (const float*)d_in[8];
    const float* adw_b  = (const float*)d_in[9];
    const float* apw_w  = (const float*)d_in[10];
    const float* apw_b  = (const float*)d_in[11];
    const float* v_w    = (const float*)d_in[12];
    const float* v_b    = (const float*)d_in[13];
    const float* po_w   = (const float*)d_in[14];
    const float* po_b   = (const float*)d_in[15];
    const float* temp_a = (const float*)d_in[16];
    const float* temp_v = (const float*)d_in[17];
    const float* ln2_w  = (const float*)d_in[18];
    const float* ln2_b  = (const float*)d_in[19];
    const float* pw1_w  = (const float*)d_in[20];
    const float* pw1_b  = (const float*)d_in[21];
    const float* dw_w   = (const float*)d_in[22];
    const float* dw_b   = (const float*)d_in[23];
    const float* pw2_w  = (const float*)d_in[24];
    const float* pw2_b  = (const float*)d_in[25];

    // ---- workspace bump allocator ----
    char* ws = (char*)d_ws;
    size_t off = 0;
    auto alloc = [&](size_t bytes) -> void* {
        off = (off + 255) & ~(size_t)255;
        void* p = ws + off;
        off += bytes;
        return p;
    };
    const size_t N1 = (size_t)B * C * HW;
    const size_t N2 = (size_t)B * C * P2;
    const size_t Na = (size_t)B * 64 * P2;
    const size_t NY = (size_t)B * MID * HW;

    __bf16* xn1_bf  = (__bf16*)alloc(N1 * 2);                 // [b][p][C] NHWC
    __bf16* kw_bf   = (__bf16*)alloc((size_t)C * C * 9 * 2);  // [co][tap][ci]
    __bf16* vw_bf   = (__bf16*)alloc((size_t)C * C * 2);
    __bf16* apw_bf  = (__bf16*)alloc((size_t)64 * C * 2);
    __bf16* pow_bf  = (__bf16*)alloc((size_t)C * C * 2);
    __bf16* w1_bf   = (__bf16*)alloc((size_t)MID * C * 2);
    __bf16* w2_bf   = (__bf16*)alloc((size_t)C * MID * 2);
    float*  q_f     = (float*)alloc(N2 * 4);                  // [b][c][p2] channel-major
    float*  k_f     = (float*)alloc(N2 * 4);                  // channel-major
    __bf16* adw_bf  = (__bf16*)alloc(N2 * 2);                 // [b][p2][C] NHWC
    float*  a_f     = (float*)alloc(Na * 4);                  // channel-major
    float*  v_f     = (float*)alloc(N1 * 4);                  // [b][p][C] NHWC
    float*  invq    = (float*)alloc(256 * 4);
    float*  invk    = (float*)alloc(256 * 4);
    float*  inva    = (float*)alloc(128 * 4);
    float*  Mmat    = (float*)alloc((size_t)B * 8 * 256 * 4);
    __bf16* ao_bf   = (__bf16*)alloc(N1 * 2);                 // NHWC
    float*  x2_f    = (float*)alloc(N1 * 4);                  // NHWC
    __bf16* xn2_bf  = (__bf16*)alloc(N1 * 2);                 // NHWC
    __bf16* y_bf    = (__bf16*)alloc(NY * 2);                 // NHWC
    __bf16* ys_bf   = (__bf16*)alloc(NY * 2);                 // NHWC
    float*  out_f   = (float*)d_out;                          // NCHW (final)

    // ---- 1. weights to bf16 (+ tap-major permute for k_w) ----
    auto cvt = [&](const float* in, __bf16* o, int n) {
        cvt_f32_bf16_kernel<<<(n + 255) / 256, 256, 0, stream>>>(in, o, n);
    };
    permute_kw_kernel<<<(C * C * 9 + 255) / 256, 256, 0, stream>>>(k_w, kw_bf, C);
    cvt(v_w,   vw_bf,  C * C);
    cvt(apw_w, apw_bf, 64 * C);
    cvt(po_w,  pow_bf, C * C);
    cvt(pw1_w, w1_bf,  MID * C);
    cvt(pw2_w, w2_bf,  C * MID);

    // ---- 2. LN1 (NCHW -> NHWC bf16) ----
    ln1_kernel<<<dim3(HW / 256, 1, B), 256, 0, stream>>>(x, ln1_w, ln1_b, xn1_bf, C, HW);

    // ---- 3. V = (Wv @ xn + b) * illu  (WMMA, NHWC out) ----
    gemm_bf16_kernel<4, 2, 1><<<dim3(HW / 256, C / 64, B), 256, 0, stream>>>(
        vw_bf, xn1_bf, v_b, v_f, nullptr, illu, C, HW, C);

    // ---- 4. K = dense 3x3 s2 conv (implicit GEMM, WMMA, NCHW out) ----
    conv3x3s2_wmma_kernel<4, 2><<<dim3(P2 / 256, C / 64, B), 256, 0, stream>>>(
        kw_bf, xn1_bf, k_b, k_f, C, C, H, W, H2, W2);

    // ---- 5. Q = depthwise 3x3 s2 (NHWC in, channel-major f32 out) ----
    dwconv3x3s2_kernel<<<dim3(P2, 1, B), C, 0, stream>>>(
        xn1_bf, q_w, q_b, q_f, nullptr, C, H, W, H2, W2);

    // ---- 6. A_dw = depthwise 3x3 s2 (NHWC bf16 out) ----
    dwconv3x3s2_kernel<<<dim3(P2, 1, B), C, 0, stream>>>(
        xn1_bf, adw_w, adw_b, nullptr, adw_bf, C, H, W, H2, W2);

    // ---- 7. A = 1x1 conv C->64 (WMMA, channel-major f32 out) ----
    gemm_bf16_kernel<4, 2, 0><<<dim3(P2 / 256, 1, B), 256, 0, stream>>>(
        apw_bf, adw_bf, apw_b, a_f, nullptr, nullptr, 64, P2, C);

    // ---- 8. row inverse L2 norms ----
    rownorm_kernel<<<B * C, 256, 0, stream>>>(q_f, invq, P2);
    rownorm_kernel<<<B * C, 256, 0, stream>>>(k_f, invk, P2);
    rownorm_kernel<<<B * 64, 256, 0, stream>>>(a_f, inva, P2);

    // ---- 9. Gram + softmax + M ----
    attn_gram_kernel<<<dim3(1, 8, B), 256, 0, stream>>>(
        q_f, k_f, a_f, invq, invk, inva, temp_a, temp_v, Mmat, P2);

    // ---- 10. out = M @ v (NHWC bf16) ----
    attn_apply_kernel<<<dim3(HW / 256, 8, B), 256, 0, stream>>>(Mmat, v_f, ao_bf, HW);

    // ---- 11. x2 = x + po(out)  (WMMA + NCHW residual, NHWC out) ----
    gemm_bf16_kernel<4, 2, 2><<<dim3(HW / 256, C / 64, B), 256, 0, stream>>>(
        pow_bf, ao_bf, po_b, x2_f, nullptr, x, C, HW, C);

    // ---- 12. LN2 (NHWC -> NHWC bf16) ----
    ln2_kernel<<<dim3(HW / 256, 1, B), 256, 0, stream>>>(x2_f, ln2_w, ln2_b, xn2_bf, C, HW);

    // ---- 13. y = pw1(xn2)  (WMMA, NHWC bf16 out) ----
    gemm_bf16_kernel<4, 2, 3><<<dim3(HW / 256, MID / 64, B), 256, 0, stream>>>(
        w1_bf, xn2_bf, pw1_b, nullptr, y_bf, nullptr, MID, HW, C);

    // ---- 14. ys = silu(dw3x3(y)) (NHWC) ----
    dwconv3x3_silu_kernel<<<dim3(HW, 1, B), MID, 0, stream>>>(
        y_bf, dw_w, dw_b, ys_bf, MID, H, W);

    // ---- 15. out = x2 + pw2(ys)  (WMMA, NCHW final out) ----
    gemm_bf16_kernel<4, 2, 4><<<dim3(HW / 256, C / 64, B), 256, 0, stream>>>(
        w2_bf, ys_bf, pw2_b, out_f, nullptr, x2_f, C, HW, MID);
}
